// RankAwareRelationEncoder_74388833566859
// MI455X (gfx1250) — matrix operand order repaired and checked by reading, not compiled, not run on hardware
//
#include <hip/hip_runtime.h>
#include <hip/hip_bf16.h>
#include <math.h>

// ---------------------------------------------------------------------------
// RankAwareRelationEncoder, folded form (see analysis):
//   attn[h,i,j] = relu( sum_f pos(base[i,j])[f] * M[h,f] + tfull[i,h] + tfull[j,h] )
// with M = attn_W @ post_W[:, :256]  (8 x 256)
//      tfull[i,h] = 0.5*c[h] + (i>=64 ? normal_rank[i-64] . M2[h] : 0)
//      M2 = attn_W @ post_W[:, 256:], c[h] = post_b.attn_W[h] + attn_b[h]
// The 256->8 projection runs on v_wmma_f32_16x16x32_f16 (K accumulated over
// 8 chunks of 32), 16 (i,j) pairs per wave.
// ---------------------------------------------------------------------------

#define NQ 448
#define NN 512
#define ND 64            // num_denoising
#define ED 256
#define NH 8
#define NC 91
#define EPSV 1e-5f
#define PAIRS (NN*NN)            // 262144
#define TILES (PAIRS/16)         // 16384
#define OUT_ATTN (NH*NN*NN)      // 2097152

typedef __attribute__((ext_vector_type(16))) _Float16 v16h;
typedef __attribute__((ext_vector_type(8)))  float    v8f;
typedef __attribute__((ext_vector_type(8)))  unsigned int v8u;

// ---------------- workspace layout (bytes) ----------------
// pbox      : 512*4 f32            @ 0        (8 KB)   permuted boxes
// Mall      : 8*512 f32            @ 8192     (16 KB)  [M | M2] per head
// nrank     : 448*256 f32          @ 24576    (448 KB) normal_rank
// Bpack     : 2048 u32             @ 483328   (8 KB)   WMMA B fragments
// tfull     : 512*8 f32            @ 491520   (16 KB)
#define WS_PBOX   0
#define WS_MALL   8192
#define WS_NRANK  24576
#define WS_BPACK  483328
#define WS_TFULL  491520

// ============================================================
// Kernel 1: scores -> bitonic argsort (desc, stable) -> pbox + rank_indices
// ============================================================
__global__ void k_sort(const float* __restrict__ boxes,
                       const float* __restrict__ logits,
                       float* __restrict__ pbox,
                       float* __restrict__ out_rank) {
  __shared__ float key[NN];
  __shared__ int   idx[NN];
  const int tid = threadIdx.x;

  float k = -INFINITY;
  if (tid < NQ) {
    const float* row = logits + (ND + tid) * NC;
    float m = row[0];
    for (int c = 1; c < NC; ++c) m = fmaxf(m, row[c]);
    k = m;                       // sigmoid is monotone: sort by raw max logit
  }
  key[tid] = k;
  idx[tid] = tid;
  __syncthreads();

  for (unsigned bk = 2; bk <= NN; bk <<= 1) {
    for (unsigned j = bk >> 1; j > 0; j >>= 1) {
      unsigned p = tid ^ j;
      if (p > (unsigned)tid) {
        float a = key[tid], b = key[p];
        int ia = idx[tid], ib = idx[p];
        // total order: higher key first; ties -> lower index first (stable)
        bool a_first = (a > b) || (a == b && ia < ib);
        bool desc = ((tid & bk) == 0);
        if (desc ? !a_first : a_first) {
          key[tid] = b; key[p] = a;
          idx[tid] = ib; idx[p] = ia;
        }
      }
      __syncthreads();
    }
  }

  // rank_indices + permuted boxes
  if (tid < ND) {
    out_rank[tid] = (float)tid;
    for (int c = 0; c < 4; ++c) pbox[tid*4 + c] = boxes[tid*4 + c];
  }
  if (tid < NQ) {
    int src = ND + idx[tid];
    out_rank[ND + tid] = (float)src;
    for (int c = 0; c < 4; ++c) pbox[(ND + tid)*4 + c] = boxes[src*4 + c];
  }
}

// ============================================================
// Kernel 2a: normal_rank = rank_emb @ pre_W.T + pre_b   (448 x 256)
// ============================================================
__global__ void k_nrank(const float* __restrict__ rank_emb,
                        const float* __restrict__ pre_W,
                        const float* __restrict__ pre_b,
                        float* __restrict__ nrank) {
  __shared__ float remb[ED];
  const int q = blockIdx.x, t = threadIdx.x;
  remb[t] = rank_emb[q*ED + t];
  __syncthreads();
  float v = pre_b[t];
  const float* wrow = pre_W + t*ED;       // pre_W[t][e]
  for (int e = 0; e < ED; ++e) v += remb[e] * wrow[e];
  nrank[q*ED + t] = v;
}

// ============================================================
// Kernel 2b: Mall[h][col] = attn_W[h] . post_W[:,col]   (8 x 512)
// ============================================================
__global__ void k_mall(const float* __restrict__ attn_W,
                       const float* __restrict__ post_W,
                       float* __restrict__ Mall) {
  const int o = blockIdx.x*256 + threadIdx.x;  // 4096 outputs
  const int h = o >> 9, col = o & 511;
  float v = 0.f;
  for (int r = 0; r < ED; ++r) v += attn_W[h*ED + r] * post_W[r*512 + col];
  Mall[o] = v;
}

// ============================================================
// Kernel 2c: pack B fragments (contiguous k order, matched by A) + tfull table
// B is (K=256) x (N=16), B[f][n] = M[n][f] for n<8 else 0.
// Lane L: N=L&15, half=L>>4; VGPR v packs K=16*half+2v (lo), +1 (hi).
// ============================================================
__global__ void k_pack(const float* __restrict__ Mall,
                       const float* __restrict__ nrank,
                       const float* __restrict__ attn_W,
                       const float* __restrict__ post_b,
                       const float* __restrict__ attn_b,
                       unsigned int* __restrict__ Bpack,
                       float* __restrict__ tfull) {
  const int id = blockIdx.x*256 + threadIdx.x;   // 6144 threads
  if (id < 2048) {
    const int ck = id >> 8, v = (id >> 5) & 7, L = id & 31;
    const int N = L & 15, hb = L >> 4;
    const int f = ck*32 + 16*hb + 2*v;           // even feature index
    unsigned lo = 0, hi = 0;
    if (N < NH) {
      _Float16 a = (_Float16)Mall[N*512 + f];
      _Float16 b = (_Float16)Mall[N*512 + f + 1];
      lo = __builtin_bit_cast(unsigned short, a);
      hi = __builtin_bit_cast(unsigned short, b);
    }
    Bpack[id] = lo | (hi << 16);
  } else {
    const int id2 = id - 2048;                   // 4096 = 512*8
    const int i = id2 >> 3, h = id2 & 7;
    float c0 = attn_b[h];
    for (int k = 0; k < ED; ++k) c0 += post_b[k] * attn_W[h*ED + k];
    float v = 0.5f * c0;
    if (i >= ND) {
      const float* nr = nrank + (i - ND)*ED;
      const float* m2 = Mall + h*512 + ED;       // M2[h]
      for (int k = 0; k < ED; ++k) v += nr[k] * m2[k];
    }
    tfull[i*NH + h] = v;
  }
}

// ============================================================
// Kernel 3: main. Each wave: 16 pairs -> pos features (f16) -> WMMA x8
// chunks against constant B (M.T) -> + tfull[i]+tfull[j] -> relu -> store.
// 128 thr/block (4 waves), 1024 blocks, 4 tiles per wave (consecutive).
// ============================================================
__global__ void __launch_bounds__(128)
k_main(const float* __restrict__ pbox,
       const unsigned int* __restrict__ Bpack,
       const float* __restrict__ tfull,
       float* __restrict__ attn) {
  __shared__ float wtab[32];                     // 100 / 10000^(m/32)
  if (threadIdx.x < 32) {
    wtab[threadIdx.x] =
        100.0f * exp2f(-(float)threadIdx.x * (13.287712379549449f / 32.0f));
  }
  __syncthreads();

  const int lane = threadIdx.x & 31;
  const int wv   = blockIdx.x * 4 + (threadIdx.x >> 5);
  const int hb   = lane >> 4;
  const int mrow = lane & 15;

  // Load constant B fragments into registers (reused across all tiles).
  v16h Bfrag[8];
#pragma unroll
  for (int ck = 0; ck < 8; ++ck) {
    v8u tmp;
#pragma unroll
    for (int v = 0; v < 8; ++v) tmp[v] = Bpack[ck*256 + v*32 + lane];
    Bfrag[ck] = __builtin_bit_cast(v16h, tmp);
  }

  for (int tt = 0; tt < 4; ++tt) {
    const int T = wv*4 + tt;                     // tile id, uniform per wave
    const int p = T*16 + mrow;
    const int i = p >> 9, j = p & 511;

    const float4 bi = ((const float4*)pbox)[i];
    const float4 bj = ((const float4*)pbox)[j];
    float base0 = logf(fabsf(bi.x - bj.x) / (bi.z + EPSV) + 1.0f);
    float base1 = logf(fabsf(bi.y - bj.y) / (bi.w + EPSV) + 1.0f);
    float base2 = logf((bi.z + EPSV) / (bj.z + EPSV));
    float base3 = logf((bi.w + EPSV) / (bj.w + EPSV));
    float baseArr[4] = {base0, base1, base2, base3};

    v8f acc = {0.f, 0.f, 0.f, 0.f, 0.f, 0.f, 0.f, 0.f};

#pragma unroll
    for (int ck = 0; ck < 8; ++ck) {
      const float bch = baseArr[ck >> 1];        // channel = ck/2 (compile-time)
      v16h A;
#pragma unroll
      for (int v = 0; v < 8; ++v) {
        // A layout (matches Bpack): lanes0-15 K=0..15; lanes16-31 K=16..31
        const int m = 16*(ck & 1) + 8*hb + v;
        float s, c;
        __sincosf(bch * wtab[m], &s, &c);
        A[2*v]     = (_Float16)s;                // even k -> sin
        A[2*v + 1] = (_Float16)c;                // odd  k -> cos
      }
      acc = __builtin_amdgcn_wmma_f32_16x16x32_f16(
          false, A, false, Bfrag[ck], (short)0, acc, false, false);
    }

    // D layout: lane L -> column N=L&15; VGPR r -> row M = r + 8*(L>>4)
    const int Ncol = lane & 15;
    if (Ncol < NH) {
#pragma unroll
      for (int r = 0; r < 8; ++r) {
        const int Mrow = r + 8*hb;
        const int pr = T*16 + Mrow;
        const int ii = pr >> 9, jj = pr & 511;
        float val = acc[r] + tfull[ii*NH + Ncol] + tfull[jj*NH + Ncol];
        attn[Ncol*PAIRS + ii*NN + jj] = fmaxf(val, 0.0f);
      }
    }
  }
}

// ============================================================
extern "C" void kernel_launch(void* const* d_in, const int* in_sizes, int n_in,
                              void* d_out, int out_size, void* d_ws, size_t ws_size,
                              hipStream_t stream) {
  (void)in_sizes; (void)n_in; (void)out_size; (void)ws_size;
  const float* src_boxes   = (const float*)d_in[0];
  const float* pred_logits = (const float*)d_in[1];
  const float* rank_emb    = (const float*)d_in[2];
  const float* pre_W       = (const float*)d_in[3];
  const float* pre_b       = (const float*)d_in[4];
  const float* post_W      = (const float*)d_in[5];
  const float* post_b      = (const float*)d_in[6];
  const float* attn_W      = (const float*)d_in[7];
  const float* attn_b      = (const float*)d_in[8];

  char* ws = (char*)d_ws;
  float*        pbox  = (float*)(ws + WS_PBOX);
  float*        Mall  = (float*)(ws + WS_MALL);
  float*        nrank = (float*)(ws + WS_NRANK);
  unsigned int* Bpack = (unsigned int*)(ws + WS_BPACK);
  float*        tfull = (float*)(ws + WS_TFULL);

  float* attn_out = (float*)d_out;               // [8,512,512]
  float* rank_out = attn_out + OUT_ATTN;         // [512]

  k_sort<<<1, 512, 0, stream>>>(src_boxes, pred_logits, pbox, rank_out);
  k_nrank<<<NQ, 256, 0, stream>>>(rank_emb, pre_W, pre_b, nrank);
  k_mall<<<16, 256, 0, stream>>>(attn_W, post_W, Mall);
  k_pack<<<24, 256, 0, stream>>>(Mall, nrank, attn_W, post_b, attn_b, Bpack, tfull);
  k_main<<<1024, 128, 0, stream>>>(pbox, Bpack, tfull, attn_out);
}
